// MultiheadAttention_85409719648595
// MI455X (gfx1250) — matrix operand order repaired and checked by reading, not compiled, not run on hardware
//
#include <hip/hip_runtime.h>
#include <hip/hip_bf16.h>

#define D_MODEL 1024
#define NHEAD   16
#define DHEAD   64
#define BATCH   4
#define SEQ     2048

typedef __attribute__((ext_vector_type(16))) _Float16 v16h;
typedef __attribute__((ext_vector_type(8)))  _Float16 v8h;
typedef __attribute__((ext_vector_type(4)))  _Float16 v4h;
typedef __attribute__((ext_vector_type(8)))  float    v8f;

#define LDA 40   // LDS stride (halves) for 32-wide K tiles, padded to dodge bank conflicts
#define LDB 40

__device__ __forceinline__ v16h cat16(v8h lo, v8h hi) {
  return __builtin_shufflevector(lo, hi, 0,1,2,3,4,5,6,7,8,9,10,11,12,13,14,15);
}

__device__ __forceinline__ v8f wmma_f16(v16h a, v16h b, v8f c) {
  return __builtin_amdgcn_wmma_f32_16x16x32_f16(false, a, false, b, (short)0, c, false, false);
}

// A operand (16x32 f16): lane L (L16=L%16,hi=L/16) holds M=L16, K = hi*8 + {0..7} and {16..23}
__device__ __forceinline__ v16h fragA(const _Float16* rowbase, int hi) {
  v8h lo = *(const v8h*)(rowbase + hi * 8);
  v8h up = *(const v8h*)(rowbase + hi * 8 + 16);
  return cat16(lo, up);
}
// B operand (32x16 f16): lane holds N=L16, K = hi*8 + {0..7} and {16..23}
__device__ __forceinline__ v16h fragB(const _Float16* nbase, int hi) {
  v8h lo = *(const v8h*)(nbase + hi * 8);
  v8h up = *(const v8h*)(nbase + hi * 8 + 16);
  return cat16(lo, up);
}

// ---------------------------------------------------------------------------
// Kernel 1: projection GEMM.  Y[b,h,s,d] = (X @ W + bias) * scale  (f16 out)
// X fp32 [8192,1024], W fp32 [1024,1024].  Tile: BM=128, BN=64, BK=32.
// ---------------------------------------------------------------------------
__global__ __launch_bounds__(256) void proj_gemm_kernel(
    const float* __restrict__ X, const float* __restrict__ W,
    const float* __restrict__ bias, _Float16* __restrict__ Y, float scale)
{
  __shared__ _Float16 la[128 * LDA];
  __shared__ _Float16 lb[64 * LDB];
  const int tid = threadIdx.x, lane = tid & 31, wv = tid >> 5;
  const int L16 = lane & 15, hi = lane >> 4;
  const int wm = wv & 3, wn = wv >> 2;          // 4x2 wave grid
  const int m0 = blockIdx.x * 128;
  const int n0 = blockIdx.y * 64;

  v8f acc[2][2] = {};
  for (int k0 = 0; k0 < D_MODEL; k0 += 32) {
    __syncthreads();
    // stage A: 128x32 fp32 -> f16, row-major (K contiguous)
#pragma unroll
    for (int i = 0; i < 4; ++i) {
      const int idx = tid + i * 256;            // 1024 float4 chunks
      const int row = idx >> 3, c4 = idx & 7;
      const float4 f = *(const float4*)(X + (size_t)(m0 + row) * D_MODEL + k0 + c4 * 4);
      v4h h4; h4[0] = (_Float16)f.x; h4[1] = (_Float16)f.y;
      h4[2] = (_Float16)f.z; h4[3] = (_Float16)f.w;
      *(v4h*)(la + row * LDA + c4 * 4) = h4;
    }
    // stage B transposed: W[k][n] -> lb[n][k]  (f16)
#pragma unroll
    for (int i = 0; i < 2; ++i) {
      const int idx = tid + i * 256;            // 512 float4 chunks
      const int kk = idx >> 4, c4 = idx & 15;
      const float4 f = *(const float4*)(W + (size_t)(k0 + kk) * D_MODEL + n0 + c4 * 4);
      lb[(c4 * 4 + 0) * LDB + kk] = (_Float16)f.x;
      lb[(c4 * 4 + 1) * LDB + kk] = (_Float16)f.y;
      lb[(c4 * 4 + 2) * LDB + kk] = (_Float16)f.z;
      lb[(c4 * 4 + 3) * LDB + kk] = (_Float16)f.w;
    }
    __syncthreads();
#pragma unroll
    for (int mt = 0; mt < 2; ++mt) {
      const v16h a = fragA(la + (wm * 32 + mt * 16 + L16) * LDA, hi);
#pragma unroll
      for (int nt = 0; nt < 2; ++nt) {
        const v16h b = fragB(lb + (wn * 32 + nt * 16 + L16) * LDB, hi);
        acc[mt][nt] = wmma_f16(a, b, acc[mt][nt]);
      }
    }
  }
  // epilogue: bias, scale, f16 store to head-major [B,H,S,64]
#pragma unroll
  for (int mt = 0; mt < 2; ++mt)
#pragma unroll
    for (int nt = 0; nt < 2; ++nt) {
      const int gn = n0 + wn * 32 + nt * 16 + L16;
      const int h = gn >> 6, d = gn & 63;
      const float bv = bias[gn];
#pragma unroll
      for (int r = 0; r < 8; ++r) {
        const int gm = m0 + wm * 32 + mt * 16 + r + hi * 8;
        const int bb = gm >> 11, ss = gm & (SEQ - 1);
        Y[(((size_t)bb * NHEAD + h) * SEQ + ss) * DHEAD + d] =
            (_Float16)((acc[mt][nt][r] + bv) * scale);
      }
    }
}

// ---------------------------------------------------------------------------
// Kernel 2: raw scores = Qh @ Kh^T per (b,h).  Both operands direct from
// global (head-major layout makes every WMMA operand 16B-contiguous).
// ---------------------------------------------------------------------------
__global__ __launch_bounds__(256) void scores_kernel(
    const _Float16* __restrict__ Qh, const _Float16* __restrict__ Kh,
    float* __restrict__ attn)
{
  const int tid = threadIdx.x, lane = tid & 31, wv = tid >> 5;
  const int L16 = lane & 15, hi = lane >> 4;
  const int bh = blockIdx.y;
  const int q0 = blockIdx.x * 128 + wv * 16;

  const _Float16* Qp = Qh + ((size_t)bh * SEQ + q0 + L16) * DHEAD;
  const v16h a0 = cat16(*(const v8h*)(Qp + hi * 8),      *(const v8h*)(Qp + hi * 8 + 16));
  const v16h a1 = cat16(*(const v8h*)(Qp + 32 + hi * 8), *(const v8h*)(Qp + 32 + hi * 8 + 16));

  const _Float16* Kbase = Kh + (size_t)bh * SEQ * DHEAD;
  float* arow = attn + ((size_t)bh * SEQ + q0) * SEQ;

  for (int n0 = 0; n0 < SEQ; n0 += 16) {
    const _Float16* Kp = Kbase + (size_t)(n0 + L16) * DHEAD;
    const v16h b0 = cat16(*(const v8h*)(Kp + hi * 8),      *(const v8h*)(Kp + hi * 8 + 16));
    const v16h b1 = cat16(*(const v8h*)(Kp + 32 + hi * 8), *(const v8h*)(Kp + 32 + hi * 8 + 16));
    v8f acc = {};
    acc = wmma_f16(a0, b0, acc);
    acc = wmma_f16(a1, b1, acc);
#pragma unroll
    for (int r = 0; r < 8; ++r)
      arow[(size_t)(r + hi * 8) * SEQ + n0 + L16] = acc[r];
  }
}

// ---------------------------------------------------------------------------
// Kernel 3: in-place row softmax over attn rows of length SEQ.
// One block per row; 256 threads x 8 elements.
// ---------------------------------------------------------------------------
__global__ __launch_bounds__(256) void softmax_kernel(float* __restrict__ attn)
{
  __shared__ float red[256];
  const int tid = threadIdx.x;
  float* row = attn + (size_t)blockIdx.x * SEQ;
  float4 x0 = *(const float4*)(row + tid * 8);
  float4 x1 = *(const float4*)(row + tid * 8 + 4);
  float m = fmaxf(fmaxf(fmaxf(x0.x, x0.y), fmaxf(x0.z, x0.w)),
                  fmaxf(fmaxf(x1.x, x1.y), fmaxf(x1.z, x1.w)));
  red[tid] = m; __syncthreads();
  for (int s2 = 128; s2 > 0; s2 >>= 1) {
    if (tid < s2) red[tid] = fmaxf(red[tid], red[tid + s2]);
    __syncthreads();
  }
  m = red[0]; __syncthreads();
  x0.x = __expf(x0.x - m); x0.y = __expf(x0.y - m);
  x0.z = __expf(x0.z - m); x0.w = __expf(x0.w - m);
  x1.x = __expf(x1.x - m); x1.y = __expf(x1.y - m);
  x1.z = __expf(x1.z - m); x1.w = __expf(x1.w - m);
  red[tid] = x0.x + x0.y + x0.z + x0.w + x1.x + x1.y + x1.z + x1.w;
  __syncthreads();
  for (int s2 = 128; s2 > 0; s2 >>= 1) {
    if (tid < s2) red[tid] += red[tid + s2];
    __syncthreads();
  }
  const float inv = 1.0f / red[0];
  x0.x *= inv; x0.y *= inv; x0.z *= inv; x0.w *= inv;
  x1.x *= inv; x1.y *= inv; x1.z *= inv; x1.w *= inv;
  *(float4*)(row + tid * 8)     = x0;
  *(float4*)(row + tid * 8 + 4) = x1;
}

// ---------------------------------------------------------------------------
// Kernel 4: ctx = attn @ Vh per (b,h).  attn fp32 -> f16 A operand on the fly;
// V tile transposed into LDS for the B operand.  ctx stored [B,S,H*64] f16.
// ---------------------------------------------------------------------------
__global__ __launch_bounds__(256) void ctx_kernel(
    const float* __restrict__ attn, const _Float16* __restrict__ Vh,
    _Float16* __restrict__ CTX)
{
  __shared__ _Float16 vt[64 * LDB];   // [d][k] transposed V tile
  const int tid = threadIdx.x, lane = tid & 31, wv = tid >> 5;
  const int L16 = lane & 15, hi = lane >> 4;
  const int bh = blockIdx.y, b = bh >> 4, h = bh & 15;
  const int q0 = blockIdx.x * 128 + wv * 16;

  const float* arow = attn + ((size_t)bh * SEQ + q0 + L16) * SEQ;
  const _Float16* Vbase = Vh + (size_t)bh * SEQ * DHEAD;
  v8f acc[4] = {};

  for (int k0 = 0; k0 < SEQ; k0 += 32) {
    __syncthreads();
    {
      const int kk = tid >> 3, c8 = tid & 7;    // 32 rows x 8 chunks of 8 halves
      v8h v = *(const v8h*)(Vbase + (size_t)(k0 + kk) * DHEAD + c8 * 8);
#pragma unroll
      for (int j = 0; j < 8; ++j) vt[(c8 * 8 + j) * LDB + kk] = v[j];
    }
    __syncthreads();
    const float* ap = arow + k0 + hi * 8;
    const float4 f0 = *(const float4*)(ap);
    const float4 f1 = *(const float4*)(ap + 4);
    const float4 f2 = *(const float4*)(ap + 16);
    const float4 f3 = *(const float4*)(ap + 20);
    v16h a;
    a[0]  = (_Float16)f0.x; a[1]  = (_Float16)f0.y; a[2]  = (_Float16)f0.z; a[3]  = (_Float16)f0.w;
    a[4]  = (_Float16)f1.x; a[5]  = (_Float16)f1.y; a[6]  = (_Float16)f1.z; a[7]  = (_Float16)f1.w;
    a[8]  = (_Float16)f2.x; a[9]  = (_Float16)f2.y; a[10] = (_Float16)f2.z; a[11] = (_Float16)f2.w;
    a[12] = (_Float16)f3.x; a[13] = (_Float16)f3.y; a[14] = (_Float16)f3.z; a[15] = (_Float16)f3.w;
#pragma unroll
    for (int nt = 0; nt < 4; ++nt) {
      const v16h bf = fragB(vt + (nt * 16 + L16) * LDB, hi);
      acc[nt] = wmma_f16(a, bf, acc[nt]);
    }
  }
#pragma unroll
  for (int nt = 0; nt < 4; ++nt)
#pragma unroll
    for (int r = 0; r < 8; ++r) {
      const int qr = q0 + r + hi * 8;
      CTX[((size_t)b * SEQ + qr) * D_MODEL + h * DHEAD + nt * 16 + L16] =
          (_Float16)acc[nt][r];
    }
}

// ---------------------------------------------------------------------------
// Kernel 5: out = ctx(f16) @ Wo + bo  -> fp32.  Same tiling as proj_gemm.
// ---------------------------------------------------------------------------
__global__ __launch_bounds__(256) void out_gemm_kernel(
    const _Float16* __restrict__ A, const float* __restrict__ W,
    const float* __restrict__ bias, float* __restrict__ Out)
{
  __shared__ _Float16 la[128 * LDA];
  __shared__ _Float16 lb[64 * LDB];
  const int tid = threadIdx.x, lane = tid & 31, wv = tid >> 5;
  const int L16 = lane & 15, hi = lane >> 4;
  const int wm = wv & 3, wn = wv >> 2;
  const int m0 = blockIdx.x * 128;
  const int n0 = blockIdx.y * 64;

  v8f acc[2][2] = {};
  for (int k0 = 0; k0 < D_MODEL; k0 += 32) {
    __syncthreads();
#pragma unroll
    for (int i = 0; i < 2; ++i) {
      const int idx = tid + i * 256;            // 512 v8h chunks of A tile
      const int row = idx >> 2, c8 = idx & 3;
      *(v8h*)(la + row * LDA + c8 * 8) =
          *(const v8h*)(A + (size_t)(m0 + row) * D_MODEL + k0 + c8 * 8);
    }
#pragma unroll
    for (int i = 0; i < 2; ++i) {
      const int idx = tid + i * 256;
      const int kk = idx >> 4, c4 = idx & 15;
      const float4 f = *(const float4*)(W + (size_t)(k0 + kk) * D_MODEL + n0 + c4 * 4);
      lb[(c4 * 4 + 0) * LDB + kk] = (_Float16)f.x;
      lb[(c4 * 4 + 1) * LDB + kk] = (_Float16)f.y;
      lb[(c4 * 4 + 2) * LDB + kk] = (_Float16)f.z;
      lb[(c4 * 4 + 3) * LDB + kk] = (_Float16)f.w;
    }
    __syncthreads();
#pragma unroll
    for (int mt = 0; mt < 2; ++mt) {
      const v16h a = fragA(la + (wm * 32 + mt * 16 + L16) * LDA, hi);
#pragma unroll
      for (int nt = 0; nt < 2; ++nt) {
        const v16h b = fragB(lb + (wn * 32 + nt * 16 + L16) * LDB, hi);
        acc[mt][nt] = wmma_f16(a, b, acc[mt][nt]);
      }
    }
  }
#pragma unroll
  for (int mt = 0; mt < 2; ++mt)
#pragma unroll
    for (int nt = 0; nt < 2; ++nt) {
      const int gn = n0 + wn * 32 + nt * 16 + L16;
      const float bv = bias[gn];
#pragma unroll
      for (int r = 0; r < 8; ++r) {
        const int gm = m0 + wm * 32 + mt * 16 + r + hi * 8;
        Out[(size_t)gm * D_MODEL + gn] = acc[mt][nt][r] + bv;
      }
    }
}

// ---------------------------------------------------------------------------
extern "C" void kernel_launch(void* const* d_in, const int* in_sizes, int n_in,
                              void* d_out, int out_size, void* d_ws, size_t ws_size,
                              hipStream_t stream) {
  const float* q  = (const float*)d_in[0];
  const float* k  = (const float*)d_in[1];
  const float* v  = (const float*)d_in[2];
  const float* Wq = (const float*)d_in[3];
  const float* bq = (const float*)d_in[4];
  const float* Wk = (const float*)d_in[5];
  const float* bk = (const float*)d_in[6];
  const float* Wv = (const float*)d_in[7];
  const float* bv = (const float*)d_in[8];
  const float* Wo = (const float*)d_in[9];
  const float* bo = (const float*)d_in[10];

  float* out  = (float*)d_out;                           // [B,S,1024]
  float* attn = out + (size_t)BATCH * SEQ * D_MODEL;     // [B,H,S,S]

  const size_t headElems = (size_t)BATCH * NHEAD * SEQ * DHEAD; // 8,388,608
  _Float16* Qh  = (_Float16*)d_ws;
  _Float16* Kh  = Qh + headElems;
  _Float16* Vh  = Kh + headElems;
  _Float16* CTX = Vh + headElems;                        // [B,S,1024] f16

  const dim3 blk(256);
  const dim3 gProj(64, 16);     // 8192/128 x 1024/64
  const dim3 gAttn(16, 64);     // 2048/128 x (B*H)

  // projections (Q scaled by 1/sqrt(DHEAD) after bias, as in reference)
  proj_gemm_kernel<<<gProj, blk, 0, stream>>>(q, Wq, bq, Qh, 0.125f);
  proj_gemm_kernel<<<gProj, blk, 0, stream>>>(k, Wk, bk, Kh, 1.0f);
  proj_gemm_kernel<<<gProj, blk, 0, stream>>>(v, Wv, bv, Vh, 1.0f);
  // raw scores -> attn region
  scores_kernel<<<gAttn, blk, 0, stream>>>(Qh, Kh, attn);
  // softmax in place (one block per row)
  softmax_kernel<<<dim3(BATCH * NHEAD * SEQ), blk, 0, stream>>>(attn);
  // context = attn @ V
  ctx_kernel<<<gAttn, blk, 0, stream>>>(attn, Vh, CTX);
  // output projection
  out_gemm_kernel<<<gProj, blk, 0, stream>>>(CTX, Wo, bo, out);
}
